// DilatedGraphConvolutionCell_54039278518416
// MI455X (gfx1250) — matrix operand order repaired and checked by reading, not compiled, not run on hardware
//
#include <hip/hip_runtime.h>
#include <hip/hip_bf16.h>

// ---------------------------------------------------------------------------
// Fixed problem dimensions (from the reference)
// ---------------------------------------------------------------------------
#define LBK    33     // LOOKBACK
#define NND    500    // N_NODES
#define FE     128    // N_FEATURES
#define DD     64     // N_DATA_FEATURES
#define DOUT   64     // N_OUTPUT_FEATURES
#define KSZ    3      // KERNEL_SIZE
#define TFEAT  36
#define FC1N   1024
#define FC2N   1024
#define SLD    512    // padded leading dim for S and layer_initial (16B-aligned rows)

typedef __attribute__((ext_vector_type(16))) __bf16 v16bf;
typedef __attribute__((ext_vector_type(8)))  __bf16 v8bf;
typedef __attribute__((ext_vector_type(8)))  float  v8f;

// f32 -> bf16 round-to-nearest-even
__device__ __forceinline__ __bf16 f2bf(float f) {
    unsigned u = __builtin_bit_cast(unsigned, f);
    unsigned short h = (unsigned short)((u + 0x7FFFu + ((u >> 16) & 1u)) >> 16);
    return __builtin_bit_cast(__bf16, h);
}
// bf16 -> f32 (exact widening)
__device__ __forceinline__ float bf2f(__bf16 b) {
    unsigned short h = __builtin_bit_cast(unsigned short, b);
    return __builtin_bit_cast(float, (unsigned)h << 16);
}
__device__ __forceinline__ __bf16 bfzero() {
    return __builtin_bit_cast(__bf16, (unsigned short)0);
}

// ---------------------------------------------------------------------------
// Batched WMMA GEMM: C[b] = act( A[b] (MxK) * B[b] (KxN, or NxK when TRANS_B) + bias )
// A and C are bf16 in memory; B is f32 (weights in d_in) when B_F32, else bf16.
// f32 accumulation via v_wmma_f32_16x16x32_bf16.
//
// Block = 128 threads = 4 waves; computes a 64(M) x 32(N) C tile (each wave a
// 16x32 strip = two WMMA tiles). The 32(K)x32(N) B panel is staged through LDS
// as bf16 [n][k]. Per-iteration addressing is strength-reduced to offset
// increments; loads stay INDEXED off the kernarg base pointers so LLVM keeps
// them on the global (not flat) path. Out-of-range rows/cols are clamped on
// load (their C entries are never stored), so the main K loop is branch-free;
// K%32 is handled by one guarded zero-filled step.
// grid = (ceil(N/32), ceil(M/64), batches); lda/ldb/ldc in elements.
// ---------------------------------------------------------------------------
template<bool TRANS_B, bool B_F32, bool BIAS, bool RELU, bool ACCUM>
__global__ void wmma_gemm(const __bf16* __restrict__ A,
                          const void*  __restrict__ Bv,
                          const float* __restrict__ bias,
                          __bf16* __restrict__ C,
                          int M, int N, int K,
                          int lda, int ldb, int ldc,
                          long strideA, long strideB, long strideC)
{
    // [n][k] bf16 panel; row stride 40 bf16 = 80B (16B aligned, conflict-free)
    __shared__ __bf16 ldsB[32][40];

    const float*  Bf = (const float*)Bv;
    const __bf16* Bh = (const __bf16*)Bv;

    const int tid  = threadIdx.x;
    const int lane = tid & 31;
    const int wave = tid >> 5;
    const int n0   = blockIdx.x * 32;
    const int m0   = (blockIdx.y * 4 + wave) * 16;

    A  += (long)blockIdx.z * strideA;
    Bf += (long)blockIdx.z * strideB;
    Bh += (long)blockIdx.z * strideB;
    C  += (long)blockIdx.z * strideC;

    const int lan15  = lane & 15;
    const int laneHi = lane >> 4;

    const int     mRow = min(m0 + lan15, M - 1);     // clamped: OOB rows never stored
    const __bf16* Arow = A + (long)mRow * lda;

    // A: two complementary 8-element runs per lane, advancing 32 elems per step.
    const __bf16* ap = Arow + laneHi * 8;

    // NN panel: 8 element offsets (k = tid>>5 + 4p), each advancing 32*ldb/step.
    // Kept as indices off Bf/Bh so the loads stay global (not flat).
    const int  nnN   = tid & 31;
    const int  nnCol = min(n0 + nnN, N - 1);
    const int  kIdx  = tid >> 5;
    const long ldbl  = (long)ldb;
    long bOff[8];
    #pragma unroll
    for (int p = 0; p < 8; ++p)
        bOff[p] = (long)(kIdx + p * 4) * ldbl + nnCol;
    const long kStepB = ldbl * 32;

    // NT panel: per-thread row base; index by kb (SGPR) inside the loop.
    const int     ntN    = tid >> 2;
    const int     ntKq   = (tid & 3) * 8;
    const __bf16* ntBase = Bh + (long)min(n0 + ntN, N - 1) * ldbl + ntKq;

    const int Kmain = K & ~31;
    v8f acc0 = {};
    v8f acc1 = {};

    for (int kb = 0; kb < Kmain; kb += 32) {
        __syncthreads();                             // protect prev iter's LDS reads
        if (TRANS_B) {
            *(v8bf*)&ldsB[ntN][ntKq] = *(const v8bf*)(ntBase + kb);
        } else {
            #pragma unroll
            for (int p = 0; p < 8; ++p) {
                if (B_F32) ldsB[nnN][kIdx + p * 4] = f2bf(Bf[bOff[p]]);
                else       ldsB[nnN][kIdx + p * 4] =      Bh[bOff[p]];
                bOff[p] += kStepB;
            }
        }
        __syncthreads();

        if (kb + 32 < Kmain)                         // next A chunk (same row: safe)
            __builtin_prefetch(ap + 32, 0, 3);

        // ---- A operand (ISA layout): VGPRs 0-3 hold k = laneHi*8 + 0..7,
        //      VGPRs 4-7 hold k = 16 + laneHi*8 + 0..7. Two 16B loads, no ALU.
        const v8bf alo = *(const v8bf*)(ap);
        const v8bf ahi = *(const v8bf*)(ap + 16);
        ap += 32;
        const v16bf a = __builtin_shufflevector(alo, ahi,
                            0, 1, 2, 3, 4, 5, 6, 7, 8, 9, 10, 11, 12, 13, 14, 15);

        // ---- B operands from LDS: 2 x 16B per N-subtile, no ALU
        const v8bf b0lo = *(const v8bf*)&ldsB[lan15][laneHi * 8];
        const v8bf b0hi = *(const v8bf*)&ldsB[lan15][laneHi * 8 + 16];
        const v8bf b1lo = *(const v8bf*)&ldsB[lan15 + 16][laneHi * 8];
        const v8bf b1hi = *(const v8bf*)&ldsB[lan15 + 16][laneHi * 8 + 16];
        const v16bf b0 = __builtin_shufflevector(b0lo, b0hi,
                            0, 1, 2, 3, 4, 5, 6, 7, 8, 9, 10, 11, 12, 13, 14, 15);
        const v16bf b1 = __builtin_shufflevector(b1lo, b1hi,
                            0, 1, 2, 3, 4, 5, 6, 7, 8, 9, 10, 11, 12, 13, 14, 15);

        acc0 = __builtin_amdgcn_wmma_f32_16x16x32_bf16(false, a, false, b0,
                                                       (short)0, acc0, false, false);
        acc1 = __builtin_amdgcn_wmma_f32_16x16x32_bf16(false, a, false, b1,
                                                       (short)0, acc1, false, false);
    }

    // ---- K tail (K % 32 != 0): one guarded, zero-filled step
    if (Kmain < K) {
        __syncthreads();
        if (TRANS_B) {
            #pragma unroll
            for (int j = 0; j < 8; ++j) {
                int k = Kmain + ntKq + j;
                ldsB[ntN][ntKq + j] = (k < K) ? ntBase[Kmain + j] : bfzero();
            }
        } else {
            #pragma unroll
            for (int p = 0; p < 8; ++p) {
                int k = kIdx + p * 4;
                bool ok = (Kmain + k < K);
                if (B_F32) ldsB[nnN][k] = ok ? f2bf(Bf[bOff[p]]) : bfzero();
                else       ldsB[nnN][k] = ok ?      Bh[bOff[p]]  : bfzero();
            }
        }
        __syncthreads();

        v16bf a;
        #pragma unroll
        for (int v = 0; v < 8; ++v) {
            int k0 = Kmain + ((v & 4) ? 16 : 0) + laneHi * 8 + (v & 3) * 2;
            a[2 * v]     = (k0     < K) ? Arow[k0]     : bfzero();
            a[2 * v + 1] = (k0 + 1 < K) ? Arow[k0 + 1] : bfzero();
        }
        const v8bf b0lo = *(const v8bf*)&ldsB[lan15][laneHi * 8];
        const v8bf b0hi = *(const v8bf*)&ldsB[lan15][laneHi * 8 + 16];
        const v8bf b1lo = *(const v8bf*)&ldsB[lan15 + 16][laneHi * 8];
        const v8bf b1hi = *(const v8bf*)&ldsB[lan15 + 16][laneHi * 8 + 16];
        const v16bf b0 = __builtin_shufflevector(b0lo, b0hi,
                            0, 1, 2, 3, 4, 5, 6, 7, 8, 9, 10, 11, 12, 13, 14, 15);
        const v16bf b1 = __builtin_shufflevector(b1lo, b1hi,
                            0, 1, 2, 3, 4, 5, 6, 7, 8, 9, 10, 11, 12, 13, 14, 15);

        acc0 = __builtin_amdgcn_wmma_f32_16x16x32_bf16(false, a, false, b0,
                                                       (short)0, acc0, false, false);
        acc1 = __builtin_amdgcn_wmma_f32_16x16x32_bf16(false, a, false, b1,
                                                       (short)0, acc1, false, false);
    }

    // ---- Store. C layout: VGPR r -> row m0+r (lanes 0-15) / m0+8+r (lanes 16-31)
    #pragma unroll
    for (int r = 0; r < 8; ++r) {
        int row = m0 + r + laneHi * 8;
        if (row >= M) continue;
        int col0 = n0 + lan15;
        int col1 = n0 + 16 + lan15;
        if (col0 < N) {
            float v = acc0[r];
            if (BIAS)  v += bias[col0];
            if (ACCUM) v += bf2f(C[(long)row * ldc + col0]);
            if (RELU)  v = v > 0.f ? v : 0.f;
            C[(long)row * ldc + col0] = f2bf(v);
        }
        if (col1 < N) {
            float v = acc1[r];
            if (BIAS)  v += bias[col1];
            if (ACCUM) v += bf2f(C[(long)row * ldc + col1]);
            if (RELU)  v = v > 0.f ? v : 0.f;
            C[(long)row * ldc + col1] = f2bf(v);
        }
    }
}

// ---------------------------------------------------------------------------
// Element-wise helpers
// ---------------------------------------------------------------------------
// f32 -> bf16 cast with zero column padding: dst (rows x dld), src (rows x scols)
__global__ void cast_pad(const float* __restrict__ src, __bf16* __restrict__ dst,
                         long rows, int scols, int dld)
{
    long i = (long)blockIdx.x * blockDim.x + threadIdx.x;
    if (i >= rows * (long)dld) return;
    int  c = (int)(i % dld);
    long r = i / dld;
    dst[i] = (c < scols) ? f2bf(src[r * (long)scols + c]) : bfzero();
}

__global__ void zero_f32(float* __restrict__ p, long n) {
    long i = (long)blockIdx.x * blockDim.x + threadIdx.x;
    if (i < n) p[i] = 0.f;
}
__global__ void zero_bf16(__bf16* __restrict__ p, long n) {
    long i = (long)blockIdx.x * blockDim.x + threadIdx.x;
    if (i < n) p[i] = bfzero();
}

// U[(n*L+l)*FE + f] += Ut[l*FE + f]   (in place on U)
__global__ void add_time_bias(__bf16* __restrict__ U, const __bf16* __restrict__ Ut,
                              long total)
{
    long i = (long)blockIdx.x * blockDim.x + threadIdx.x;
    if (i >= total) return;
    int  f = (int)(i % FE);
    long r = i / FE;
    int  l = (int)(r % LBK);
    U[i] = f2bf(bf2f(U[i]) + bf2f(Ut[(long)l * FE + f]));
}

// Gather rows by modular time index (bf16).
// dst[(t*Nn+n)*W + w] = src[srow*W + w],
//   srow = timeMajor ? st*Nn + n : n*LBK + st,  st = mod(t*dil + koff, LBK)
__global__ void gather_rows(const __bf16* __restrict__ src, __bf16* __restrict__ dst,
                            int T, int Nn, int W, int dil, int koff, int timeMajor)
{
    long i = (long)blockIdx.x * blockDim.x + threadIdx.x;
    long total = (long)T * Nn * W;
    if (i >= total) return;
    int  w = (int)(i % W);
    long r = i / W;
    int  n = (int)(r % Nn);
    int  t = (int)(r / Nn);
    int st = (t * dil + koff) % LBK;
    if (st < 0) st += LBK;
    long srow = timeMajor ? ((long)st * Nn + n) : ((long)n * LBK + st);
    dst[i] = src[srow * W + w];
}

// Threshold (>= 0.05 else 0) then row softmax on bf16 rows (stride ld).
__global__ void row_softmax_thresh(__bf16* __restrict__ S, int rowLen, int ld)
{
    __shared__ float red[256];
    __bf16* row = S + (long)blockIdx.x * ld;
    const int tid = threadIdx.x;

    float mx = -3.4e38f;
    for (int i = tid; i < rowLen; i += 256) {
        float v = bf2f(row[i]);
        v = (v >= 0.05f) ? v : 0.f;
        row[i] = f2bf(v);
        mx = fmaxf(mx, v);
    }
    red[tid] = mx; __syncthreads();
    for (int s = 128; s > 0; s >>= 1) {
        if (tid < s) red[tid] = fmaxf(red[tid], red[tid + s]);
        __syncthreads();
    }
    mx = red[0]; __syncthreads();

    float sum = 0.f;
    for (int i = tid; i < rowLen; i += 256) {
        float e = __expf(bf2f(row[i]) - mx);
        row[i] = f2bf(e);
        sum += e;
    }
    red[tid] = sum; __syncthreads();
    for (int s = 128; s > 0; s >>= 1) {
        if (tid < s) red[tid] += red[tid + s];
        __syncthreads();
    }
    float inv = 1.f / red[0];
    for (int i = tid; i < rowLen; i += 256)
        row[i] = f2bf(bf2f(row[i]) * inv);
    // Note: post-softmax degree d == 1, so d^-1/2 * A * d^1/2 == A (identity).
}

// Zacc[i] (f32) += relu(H[i] (bf16) + bconv[i % DOUT])
__global__ void acc_relu_bias(float* __restrict__ Zacc, const __bf16* __restrict__ H,
                              const float* __restrict__ bconv, long total)
{
    long i = (long)blockIdx.x * blockDim.x + threadIdx.x;
    if (i >= total) return;
    float v = bf2f(H[i]) + bconv[(int)(i % DOUT)];
    Zacc[i] += (v > 0.f) ? v : 0.f;
}

// Z[(t*dil)*N*DO + j] = bf16(Zacc[t*N*DO + j])   (Z pre-zeroed)
__global__ void scatter_time(const float* __restrict__ Zacc, __bf16* __restrict__ Z,
                             int T, int dil)
{
    long per = (long)NND * DOUT;
    long i = (long)blockIdx.x * blockDim.x + threadIdx.x;
    if (i >= (long)T * per) return;
    long j = i % per;
    int  t = (int)(i / per);
    Z[(long)(t * dil) * per + j] = f2bf(Zacc[i]);
}

// out[(lvl,n,o)] = f32(Zlvl[(L-1)*N*DO + n*DO + o])
__global__ void collect_out(const __bf16* __restrict__ Z0, const __bf16* __restrict__ Z1,
                            const __bf16* __restrict__ Z2, const __bf16* __restrict__ Z3,
                            float* __restrict__ out)
{
    const int per = NND * DOUT;
    int i = blockIdx.x * blockDim.x + threadIdx.x;
    if (i >= 4 * per) return;
    int lvl = i / per, j = i % per;
    const __bf16* Z = (lvl == 0) ? Z0 : (lvl == 1) ? Z1 : (lvl == 2) ? Z2 : Z3;
    out[i] = bf2f(Z[(long)(LBK - 1) * per + j]);
}

// ---------------------------------------------------------------------------
// Host-side helpers
// ---------------------------------------------------------------------------
// mode: 0 = NN/f32B + bias + relu, 1 = NN/f32B plain,
//       2 = NT/bf16B plain,        3 = NN/bf16B plain, 4 = NN/bf16B accumulate
static void run_gemm(hipStream_t st, const __bf16* A, const void* B, const float* bias,
                     __bf16* C, int M, int N, int K, int lda, int ldb, int ldc,
                     int batches, long sA, long sB, long sC, int mode)
{
    dim3 grid((unsigned)((N + 31) / 32), (unsigned)((M + 63) / 64), (unsigned)batches);
    dim3 blk(128, 1, 1);
    switch (mode) {
    case 0: wmma_gemm<false, true,  true,  true,  false><<<grid, blk, 0, st>>>(A, B, bias, C, M, N, K, lda, ldb, ldc, sA, sB, sC); break;
    case 1: wmma_gemm<false, true,  false, false, false><<<grid, blk, 0, st>>>(A, B, bias, C, M, N, K, lda, ldb, ldc, sA, sB, sC); break;
    case 2: wmma_gemm<true,  false, false, false, false><<<grid, blk, 0, st>>>(A, B, bias, C, M, N, K, lda, ldb, ldc, sA, sB, sC); break;
    case 3: wmma_gemm<false, false, false, false, false><<<grid, blk, 0, st>>>(A, B, bias, C, M, N, K, lda, ldb, ldc, sA, sB, sC); break;
    default:wmma_gemm<false, false, false, false, true ><<<grid, blk, 0, st>>>(A, B, bias, C, M, N, K, lda, ldb, ldc, sA, sB, sC); break;
    }
}

static void run_zero16(hipStream_t st, __bf16* p, long n)
{
    zero_bf16<<<(unsigned)((n + 255) / 256), 256, 0, st>>>(p, n);
}

// ---------------------------------------------------------------------------
// Orchestration
// ---------------------------------------------------------------------------
extern "C" void kernel_launch(void* const* d_in, const int* in_sizes, int n_in,
                              void* d_out, int out_size, void* d_ws, size_t ws_size,
                              hipStream_t stream)
{
    (void)in_sizes; (void)n_in; (void)out_size; (void)ws_size;

    // Inputs (setup_inputs order)
    const float* observation = (const float*)d_in[0];   // (33, 32000)
    const float* time_feats  = (const float*)d_in[1];   // (33, 18000)
    const float* layer_init  = (const float*)d_in[2];   // (500, 33, 500) -> (16500, 500)
    const float* Ws1  = (const float*)d_in[3];
    const float* bs1  = (const float*)d_in[4];
    const float* Ws2  = (const float*)d_in[5];
    const float* bs2  = (const float*)d_in[6];
    const float* Wt1  = (const float*)d_in[7];
    const float* bt1  = (const float*)d_in[8];
    const float* Wt2  = (const float*)d_in[9];
    const float* bt2  = (const float*)d_in[10];
    const float* Bmat = (const float*)d_in[11];
    const float* Wfc1 = (const float*)d_in[12];
    const float* bfc1 = (const float*)d_in[13];
    const float* Wfc2 = (const float*)d_in[14];
    const float* bfc2 = (const float*)d_in[15];
    const float* Wfc3 = (const float*)d_in[16];
    const float* bfc3 = (const float*)d_in[17];
    const float* Wf   = (const float*)d_in[18];         // (3, 128, 64)
    const float* Wb   = (const float*)d_in[19];         // (3, 128, 64)
    const float* bconv= (const float*)d_in[20];
    float* out = (float*)d_out;                         // (4, 500, 64)

    const long NL = (long)NND * LBK;                    // 16500

    // Workspace: f32 region first (Zacc), then bf16 pool (all multiples of 8 elems)
    float*  Zacc = (float*)d_ws;
    __bf16* hws  = (__bf16*)((char*)d_ws + (size_t)LBK * NND * DOUT * sizeof(float));
    size_t off = 0;
    auto alloc = [&](size_t ne) { __bf16* p = hws + off; off += ne; return p; };

    __bf16* LIp   = alloc((size_t)NL * SLD);            // layer_init, ld 512 (pad 0)
    __bf16* obs_b = alloc((size_t)LBK * NND * DD);      // observation bf16
    __bf16* tf_b  = alloc((size_t)LBK * NND * TFEAT);   // time_feats bf16
    __bf16* Us_h  = alloc((size_t)NL * 256);
    __bf16* U     = alloc((size_t)NL * FE);
    __bf16* UBm   = alloc((size_t)NL * FE);
    __bf16* Ut_h  = alloc((size_t)LBK * 256);
    __bf16* Ut    = alloc((size_t)LBK * FE);
    __bf16* Zl[4];
    for (int i = 0; i < 4; ++i) Zl[i] = alloc((size_t)LBK * NND * DOUT);
    __bf16* H1    = alloc((size_t)LBK * FC1N);
    __bf16* H2    = alloc((size_t)LBK * FC2N);
    __bf16* Xall  = alloc((size_t)LBK * NND * FE);
    __bf16* Xk    = alloc((size_t)LBK * NND * FE);
    __bf16* U1g   = alloc((size_t)LBK * NND * FE);
    __bf16* U2g   = alloc((size_t)LBK * NND * FE);
    __bf16* S     = alloc((size_t)LBK * NND * SLD);     // ld 512 (cols 500+ unused)
    __bf16* P     = alloc((size_t)LBK * NND * DOUT);
    __bf16* Q     = alloc((size_t)LBK * NND * DOUT);
    __bf16* Hbuf  = alloc((size_t)LBK * NND * DOUT);

    auto cast_in = [&](const float* src, __bf16* dst, long rows, int scols, int dld) {
        long total = rows * (long)dld;
        cast_pad<<<(unsigned)((total + 255) / 256), 256, 0, stream>>>(src, dst, rows, scols, dld);
    };

    // ---------------- One-time f32 -> bf16 casts of A-side inputs ----------------
    cast_in(layer_init,  LIp,   NL,  NND,         SLD);
    cast_in(observation, obs_b, LBK, NND * DD,    NND * DD);
    cast_in(time_feats,  tf_b,  LBK, NND * TFEAT, NND * TFEAT);

    // ---------------- Node embeddings ----------------
    // Us = relu(relu(LI @ Ws1 + bs1) @ Ws2 + bs2)   (16500 x 128)
    run_gemm(stream, LIp,  Ws1, bs1, Us_h, (int)NL, 256, NND, SLD, 256, 256, 1, 0, 0, 0, 0);
    run_gemm(stream, Us_h, Ws2, bs2, U,    (int)NL, FE,  256, 256, FE,  FE,  1, 0, 0, 0, 0);
    // Ut = relu(relu(TF @ Wt1 + bt1) @ Wt2 + bt2)   (33 x 128)
    run_gemm(stream, tf_b, Wt1, bt1, Ut_h, LBK, 256, NND * TFEAT, NND * TFEAT, 256, 256, 1, 0, 0, 0, 0);
    run_gemm(stream, Ut_h, Wt2, bt2, Ut,   LBK, FE,  256,         256,         FE,  FE,  1, 0, 0, 0, 0);
    // U = Us + Ut broadcast (in place)
    {
        long total = NL * FE;
        add_time_bias<<<(unsigned)((total + 255) / 256), 256, 0, stream>>>(U, Ut, total);
    }
    // UB = U @ B  (bilinear form applied once)
    run_gemm(stream, U, Bmat, nullptr, UBm, (int)NL, FE, FE, FE, FE, FE, 1, 0, 0, 0, 1);

    // ---------------- Dilated conv layers ----------------
    const int dils[4] = {1, 2, 4, 8};
    const __bf16* Xin = obs_b;                          // (33, 32000)

    for (int layer = 0; layer < 4; ++layer) {
        const int dil = dils[layer];
        const int T   = (LBK + dil - 1) / dil;          // 33, 17, 9, 5
        const long TN   = (long)T * NND;
        const long Tout = TN * DOUT;

        // fc(x): relu chain -> Xall (33, 500, 128)
        run_gemm(stream, Xin, Wfc1, bfc1, H1,   LBK, FC1N,     NND * DD, NND * DD, FC1N,     FC1N,     1, 0, 0, 0, 0);
        run_gemm(stream, H1,  Wfc2, bfc2, H2,   LBK, FC2N,     FC1N,     FC1N,     FC2N,     FC2N,     1, 0, 0, 0, 0);
        run_gemm(stream, H2,  Wfc3, bfc3, Xall, LBK, NND * FE, FC2N,     FC2N,     NND * FE, NND * FE, 1, 0, 0, 0, 0);

        zero_f32<<<(unsigned)((Tout + 255) / 256), 256, 0, stream>>>(Zacc, Tout);
        // U2 gather (i2 = ts) is shared by every k / direction
        {
            long total = TN * FE;
            gather_rows<<<(unsigned)((total + 255) / 256), 256, 0, stream>>>(
                U, U2g, T, NND, FE, dil, 0, /*timeMajor=*/0);
        }

        for (int k = 0; k < KSZ; ++k) {
            // Xk = Xall[(ts - k) % L]   (time-major gather)
            {
                long total = TN * FE;
                gather_rows<<<(unsigned)((total + 255) / 256), 256, 0, stream>>>(
                    Xall, Xk, T, NND, FE, dil, -k, /*timeMajor=*/1);
            }
            // Project features before the N x N multiply (associativity: halves cost)
            run_gemm(stream, Xk, Wf + (size_t)k * FE * DOUT, nullptr, P, (int)TN, DOUT, FE, FE, DOUT, DOUT, 1, 0, 0, 0, 1);
            run_gemm(stream, Xk, Wb + (size_t)k * FE * DOUT, nullptr, Q, (int)TN, DOUT, FE, FE, DOUT, DOUT, 1, 0, 0, 0, 1);

            for (int dir = 0; dir < 2; ++dir) {
                const int koff = (dir == 0) ? -k : k;
                // U1 gather of UB rows at i1 = (ts +/- k) % L  (node-major)
                long total = TN * FE;
                gather_rows<<<(unsigned)((total + 255) / 256), 256, 0, stream>>>(
                    UBm, U1g, T, NND, FE, dil, koff, /*timeMajor=*/0);
                // S[t] = U1g[t] (500x128) @ U2g[t]^T  -> (T, 500, 500) ld 512, batched NT
                run_gemm(stream, U1g, U2g, nullptr, S, NND, NND, FE, FE, FE, SLD, T,
                         (long)NND * FE, (long)NND * FE, (long)NND * SLD, 2);
                // threshold + softmax per row -> normalized adjacency (d == 1)
                row_softmax_thresh<<<(unsigned)(T * NND), 256, 0, stream>>>(S, NND, SLD);
                // Hbuf (=/+)= S @ (P or Q), batched 500x64x500
                run_gemm(stream, S, (dir == 0) ? P : Q, nullptr, Hbuf, NND, DOUT, NND,
                         SLD, DOUT, DOUT, T,
                         (long)NND * SLD, (long)NND * DOUT, (long)NND * DOUT,
                         (dir == 0) ? 3 : 4);
            }
            // Zacc += relu(Hbuf + bconv)
            acc_relu_bias<<<(unsigned)((Tout + 255) / 256), 256, 0, stream>>>(
                Zacc, Hbuf, bconv, Tout);
        }

        // Z[ts] = Zacc, inactive timesteps zero
        run_zero16(stream, Zl[layer], (long)LBK * NND * DOUT);
        scatter_time<<<(unsigned)((Tout + 255) / 256), 256, 0, stream>>>(
            Zacc, Zl[layer], T, dil);

        Xin = Zl[layer];                                // next layer input (33, 32000)
    }

    // stack([Z0[-1], Z1[-1], Z2[-1], Z3[-1]]) -> (4, 500, 64)
    collect_out<<<(unsigned)((4 * NND * DOUT + 255) / 256), 256, 0, stream>>>(
        Zl[0], Zl[1], Zl[2], Zl[3], out);
}
